// Seq2Seq_7404523618351
// MI455X (gfx1250) — matrix operand order repaired and checked by reading, not compiled, not run on hardware
//
#include <hip/hip_runtime.h>
#include <hip/hip_bf16.h>

#define Hh 256
#define Ss 64
#define Tt 48
#define Bb 32
#define VSs 32000
#define VTt 32000
#define G4H 1024   // 4*H

typedef __attribute__((ext_vector_type(16))) _Float16 v16h;
typedef __attribute__((ext_vector_type(8)))  _Float16 v8h;
typedef __attribute__((ext_vector_type(8)))  float    v8f;

__device__ __forceinline__ float sigf(float x) { return 1.0f / (1.0f + __expf(-x)); }

// ---------------------------------------------------------------------------
// f32 -> f16 convert (weights)
// ---------------------------------------------------------------------------
__global__ void f32_to_f16_kernel(const float* __restrict__ in, _Float16* __restrict__ out, int n) {
    int i = blockIdx.x * blockDim.x + threadIdx.x;
    if (i < n) out[i] = (_Float16)in[i];
}

// ---------------------------------------------------------------------------
// Embedding gather -> f16 rows of H=256
// ---------------------------------------------------------------------------
__global__ void embed_kernel(const int* __restrict__ tok, const float* __restrict__ emb,
                             _Float16* __restrict__ out, int rows) {
    int idx = blockIdx.x * blockDim.x + threadIdx.x;
    if (idx >= rows * Hh) return;
    int r = idx >> 8, j = idx & 255;
    out[idx] = (_Float16)emb[(size_t)tok[r] * Hh + j];
}

// ---------------------------------------------------------------------------
// Generic WMMA GEMM:  C[M x N] = A1[M x K1] * W1[N x K1]^T (+ A2*W2^T) (+ bias)
// A, W are f16 row-major with ld == K.  C is f32 row-major with ld == N.
// M % (16*MPW) == 0, N % 64 == 0, K % 32 == 0.
// One wave owns a (16*MPW) x 64 C tile: each B fragment is loaded once and
// reused MPW times, halving L2 weight traffic for MPW=2 (M=32 per-step GEMMs
// and the dominant 32x32000 logits projection).
// Emits v_wmma_f32_16x16x32_f16 with f32 accumulation.
// ---------------------------------------------------------------------------
template <int MPW>
__device__ __forceinline__ void gemm_accum_pair(const _Float16* __restrict__ A,
                                                const _Float16* __restrict__ W,
                                                int K, int mbase, int n0, int lane,
                                                v8f acc[MPW][4]) {
    const int hi = (lane >> 4) & 1;
    const int nb = n0 + (lane & 15);
    for (int k0 = 0; k0 < K; k0 += 32) {
        // A fragments (16x32): lane half 'hi' holds 16 contiguous K 16*hi..16*hi+15 (matches B)
        v16h a[MPW];
#pragma unroll
        for (int mi = 0; mi < MPW; ++mi) {
            const _Float16* ap = A + (size_t)(mbase + mi * 16 + (lane & 15)) * K + k0 + 16 * hi;
            v8h alo = *(const v8h*)(ap);
            v8h ahi = *(const v8h*)(ap + 8);
#pragma unroll
            for (int i = 0; i < 8; ++i) { a[mi][i] = alo[i]; a[mi][i + 8] = ahi[i]; }
        }
#pragma unroll
        for (int j = 0; j < 4; ++j) {
            // B fragment (32x16): lane = column n, 16 contiguous K values per lane half
            const _Float16* wp = W + (size_t)(nb + 16 * j) * K + k0 + 16 * hi;
            v16h b = *(const v16h*)wp;
#pragma unroll
            for (int mi = 0; mi < MPW; ++mi)
                acc[mi][j] = __builtin_amdgcn_wmma_f32_16x16x32_f16(
                    false, a[mi], false, b, (short)0, acc[mi][j], false, false);
        }
    }
}

template <int MPW>
__global__ void wmma_gemm_kernel(const _Float16* __restrict__ A1, const _Float16* __restrict__ W1, int K1,
                                 const _Float16* __restrict__ A2, const _Float16* __restrict__ W2, int K2,
                                 const float* __restrict__ bias,
                                 float* __restrict__ C,
                                 int mGroups, int tilesN4, int N) {
    int wave = (blockIdx.x << 3) + (threadIdx.x >> 5);
    if (wave >= mGroups * tilesN4) return;
    int lane = threadIdx.x & 31;
    int mbase = (wave / tilesN4) * (16 * MPW);
    int n0 = (wave % tilesN4) * 64;
    v8f acc[MPW][4] = {};
    gemm_accum_pair<MPW>(A1, W1, K1, mbase, n0, lane, acc);
    if (A2) gemm_accum_pair<MPW>(A2, W2, K2, mbase, n0, lane, acc);
    const int hi = (lane >> 4) & 1;
#pragma unroll
    for (int j = 0; j < 4; ++j) {
        int n = n0 + 16 * j + (lane & 15);
        float badd = bias ? bias[n] : 0.0f;
#pragma unroll
        for (int mi = 0; mi < MPW; ++mi) {
#pragma unroll
            for (int r = 0; r < 8; ++r) {
                int mm = mbase + mi * 16 + r + 8 * hi;
                C[(size_t)mm * N + n] = acc[mi][j][r] + badd;
            }
        }
    }
}

// ---------------------------------------------------------------------------
// Fused LSTM gates: g = g1 (+ g2) + bih + bhh; i,f,g,o split; mask; write
// h (f32), c (f32), h (f16 for next WMMA GEMM), optional h store (seq output).
// One thread per (b, h) of B*H = 8192.
// ---------------------------------------------------------------------------
__global__ void lstm_gates_kernel(const float* __restrict__ g1, const float* __restrict__ g2,
                                  const float* __restrict__ bih, const float* __restrict__ bhh,
                                  const int* __restrict__ msrc,
                                  float* __restrict__ h, float* __restrict__ c,
                                  _Float16* __restrict__ h16,
                                  float* __restrict__ hstore) {
    int idx = blockIdx.x * blockDim.x + threadIdx.x;   // < B*H
    int b = idx >> 8;
    int j = idx & 255;
    int base = b * G4H + j;
    float gi = g1[base]           + bih[j]           + bhh[j];
    float gf = g1[base + Hh]      + bih[j + Hh]      + bhh[j + Hh];
    float gg = g1[base + 2 * Hh]  + bih[j + 2 * Hh]  + bhh[j + 2 * Hh];
    float go = g1[base + 3 * Hh]  + bih[j + 3 * Hh]  + bhh[j + 3 * Hh];
    if (g2) {
        gi += g2[base];          gf += g2[base + Hh];
        gg += g2[base + 2 * Hh]; go += g2[base + 3 * Hh];
    }
    float oldc = c[idx], oldh = h[idx];
    float nc = sigf(gf) * oldc + sigf(gi) * tanhf(gg);
    float nh = sigf(go) * tanhf(nc);
    if (msrc && msrc[b] == 1) { nc = oldc; nh = oldh; }
    c[idx] = nc;
    h[idx] = nh;
    h16[idx] = (_Float16)nh;
    if (hstore) hstore[idx] = nh;
}

// ---------------------------------------------------------------------------
// enc_out = fh + 0.5 * bh
// ---------------------------------------------------------------------------
__global__ void combine_kernel(const float* __restrict__ fh, const float* __restrict__ bh,
                               float* __restrict__ eo, int n) {
    int i = blockIdx.x * blockDim.x + threadIdx.x;
    if (i < n) eo[i] = fh[i] + 0.5f * bh[i];
}

// ---------------------------------------------------------------------------
// Decoder input assembly: x16 = [dec_emb[trg_t], ctx]  (B x 2H, f16)
// ---------------------------------------------------------------------------
__global__ void dec_input_kernel(const int* __restrict__ trg_t, const float* __restrict__ demb,
                                 const float* __restrict__ ctx, _Float16* __restrict__ x16) {
    int idx = blockIdx.x * blockDim.x + threadIdx.x;   // B * 2H
    int b = idx >> 9, j = idx & 511;
    float v = (j < Hh) ? demb[(size_t)trg_t[b] * Hh + j] : ctx[b * Hh + (j - Hh)];
    x16[idx] = (_Float16)v;
}

// ---------------------------------------------------------------------------
// Attention: scores = enc_out[s,b,:]·h2[b,:], mask src==1 -> -700, softmax,
// ctx = sum_s w*enc_out, write attn weights, build cat16 = [h2, ctx] in f16.
// One block per batch element.
// ---------------------------------------------------------------------------
__global__ void attn_kernel(const float* __restrict__ enc_out, const float* __restrict__ h2,
                            const int* __restrict__ src,
                            float* __restrict__ attn_out, float* __restrict__ ctx,
                            _Float16* __restrict__ cat16) {
    __shared__ float sc[Ss];
    int b = blockIdx.x;
    int tid = threadIdx.x;
    if (tid < Ss) {
        int s = tid;
        const float* e = enc_out + ((size_t)s * Bb + b) * Hh;
        const float* hh = h2 + b * Hh;
        float d = 0.0f;
        for (int k = 0; k < Hh; ++k) d += e[k] * hh[k];
        if (src[s * Bb + b] == 1) d = -700.0f;
        sc[s] = d;
    }
    __syncthreads();
    if (tid == 0) {
        float mx = -1e30f;
        for (int s = 0; s < Ss; ++s) mx = fmaxf(mx, sc[s]);
        float sm = 0.0f;
        for (int s = 0; s < Ss; ++s) { float e = __expf(sc[s] - mx); sc[s] = e; sm += e; }
        float inv = 1.0f / sm;
        for (int s = 0; s < Ss; ++s) sc[s] *= inv;
    }
    __syncthreads();
    if (tid < Ss) attn_out[b * Ss + tid] = sc[tid];
    if (tid < Hh) {
        float cv = 0.0f;
        for (int s = 0; s < Ss; ++s) cv += sc[s] * enc_out[((size_t)s * Bb + b) * Hh + tid];
        ctx[b * Hh + tid] = cv;
        cat16[b * (2 * Hh) + Hh + tid] = (_Float16)cv;
        cat16[b * (2 * Hh) + tid]      = (_Float16)h2[b * Hh + tid];
    }
}

// ---------------------------------------------------------------------------
// Row argmax over VT=32000 -> labels (stored as float). One block per row.
// ---------------------------------------------------------------------------
__global__ void argmax_kernel(const float* __restrict__ logits, float* __restrict__ labels, int N) {
    __shared__ float smax[256];
    __shared__ int   sidx[256];
    int b = blockIdx.x, tid = threadIdx.x;
    const float* row = logits + (size_t)b * N;
    float best = -1e30f; int bi = 0;
    for (int i = tid; i < N; i += 256) {
        float v = row[i];
        if (v > best) { best = v; bi = i; }
    }
    smax[tid] = best; sidx[tid] = bi;
    __syncthreads();
    for (int off = 128; off > 0; off >>= 1) {
        if (tid < off) {
            if (smax[tid + off] > smax[tid] ||
                (smax[tid + off] == smax[tid] && sidx[tid + off] < sidx[tid])) {
                smax[tid] = smax[tid + off]; sidx[tid] = sidx[tid + off];
            }
        }
        __syncthreads();
    }
    if (tid == 0) labels[b] = (float)sidx[0];
}

// ---------------------------------------------------------------------------
// Host orchestration
// ---------------------------------------------------------------------------
extern "C" void kernel_launch(void* const* d_in, const int* in_sizes, int n_in,
                              void* d_out, int out_size, void* d_ws, size_t ws_size,
                              hipStream_t stream) {
    (void)in_sizes; (void)n_in; (void)out_size; (void)ws_size;

    const int*   src     = (const int*)  d_in[0];
    const int*   trg     = (const int*)  d_in[1];
    const float* enc_emb = (const float*)d_in[2];
    const float* dec_emb = (const float*)d_in[3];
    const float* f1_Wih = (const float*)d_in[4],  *f1_Whh = (const float*)d_in[5];
    const float* f1_bih = (const float*)d_in[6],  *f1_bhh = (const float*)d_in[7];
    const float* f2_Wih = (const float*)d_in[8],  *f2_Whh = (const float*)d_in[9];
    const float* f2_bih = (const float*)d_in[10], *f2_bhh = (const float*)d_in[11];
    const float* b1_Wih = (const float*)d_in[12], *b1_Whh = (const float*)d_in[13];
    const float* b1_bih = (const float*)d_in[14], *b1_bhh = (const float*)d_in[15];
    const float* b2_Wih = (const float*)d_in[16], *b2_Whh = (const float*)d_in[17];
    const float* b2_bih = (const float*)d_in[18], *b2_bhh = (const float*)d_in[19];
    const float* d1_Wih = (const float*)d_in[20], *d1_Whh = (const float*)d_in[21];
    const float* d1_bih = (const float*)d_in[22], *d1_bhh = (const float*)d_in[23];
    const float* d2_Wih = (const float*)d_in[24], *d2_Whh = (const float*)d_in[25];
    const float* d2_bih = (const float*)d_in[26], *d2_bhh = (const float*)d_in[27];
    const float* out_W  = (const float*)d_in[28];
    const float* out_b  = (const float*)d_in[29];

    float* out_f      = (float*)d_out;
    float* labels_out = out_f;                                            // T*B
    float* logits_out = out_f + (size_t)Tt * Bb;                          // T*B*VT
    float* attn_out   = out_f + (size_t)Tt * Bb + (size_t)Tt * Bb * VTt;  // T*B*S

    // ---- workspace carve (bump allocator, 256B aligned) ----
    char* p = (char*)d_ws;
    auto alloc = [&](size_t bytes) -> void* {
        void* r = (void*)p;
        p += (bytes + 255) & ~(size_t)255;
        return r;
    };
    const size_t BH = (size_t)Bb * Hh;

    _Float16* Wf1ih = (_Float16*)alloc(G4H * Hh * 2);
    _Float16* Wf1hh = (_Float16*)alloc(G4H * Hh * 2);
    _Float16* Wf2ih = (_Float16*)alloc(G4H * Hh * 2);
    _Float16* Wf2hh = (_Float16*)alloc(G4H * Hh * 2);
    _Float16* Wb1ih = (_Float16*)alloc(G4H * Hh * 2);
    _Float16* Wb1hh = (_Float16*)alloc(G4H * Hh * 2);
    _Float16* Wb2ih = (_Float16*)alloc(G4H * Hh * 2);
    _Float16* Wb2hh = (_Float16*)alloc(G4H * Hh * 2);
    _Float16* Wd1ih = (_Float16*)alloc((size_t)G4H * 2 * Hh * 2);   // 1024 x 512
    _Float16* Wd1hh = (_Float16*)alloc(G4H * Hh * 2);
    _Float16* Wd2ih = (_Float16*)alloc(G4H * Hh * 2);
    _Float16* Wd2hh = (_Float16*)alloc(G4H * Hh * 2);
    _Float16* WoutT = (_Float16*)alloc((size_t)VTt * 2 * Hh * 2);   // 32000 x 512

    _Float16* X16   = (_Float16*)alloc((size_t)Ss * Bb * Hh * 2);   // encoder embeddings
    float* GpreF    = (float*)alloc((size_t)Ss * Bb * G4H * 4);     // x@Wih.T fwd L1
    float* GpreB    = (float*)alloc((size_t)Ss * Bb * G4H * 4);     // x@Wih.T bwd L1
    float* gh1      = (float*)alloc((size_t)Bb * G4H * 4);
    float* gbuf     = (float*)alloc((size_t)Bb * G4H * 4);
    float* fh       = (float*)alloc((size_t)Ss * BH * 4);
    float* bh       = (float*)alloc((size_t)Ss * BH * 4);
    float* enc_out  = (float*)alloc((size_t)Ss * BH * 4);
    _Float16* x16   = (_Float16*)alloc(BH * 2 * 2);                 // 32 x 512
    _Float16* cat16 = (_Float16*)alloc(BH * 2 * 2);                 // 32 x 512

    // zero-initialized state region (one memset)
    char* zbeg = p;
    float* h1f = (float*)alloc(BH * 4); float* c1f = (float*)alloc(BH * 4);
    float* h2f = (float*)alloc(BH * 4); float* c2f = (float*)alloc(BH * 4);
    float* h1b = (float*)alloc(BH * 4); float* c1b = (float*)alloc(BH * 4);
    float* h2b = (float*)alloc(BH * 4); float* c2b = (float*)alloc(BH * 4);
    _Float16* h1f16 = (_Float16*)alloc(BH * 2);
    _Float16* h2f16 = (_Float16*)alloc(BH * 2);
    _Float16* h1b16 = (_Float16*)alloc(BH * 2);
    _Float16* h2b16 = (_Float16*)alloc(BH * 2);
    float* ctx = (float*)alloc(BH * 4);
    char* zend = p;
    hipMemsetAsync(zbeg, 0, (size_t)(zend - zbeg), stream);

    // ---- weight conversion f32 -> f16 ----
    auto cvt = [&](const float* s_, _Float16* d_, size_t n) {
        f32_to_f16_kernel<<<dim3((unsigned)((n + 255) / 256)), dim3(256), 0, stream>>>(s_, d_, (int)n);
    };
    const size_t WN = (size_t)G4H * Hh;
    cvt(f1_Wih, Wf1ih, WN); cvt(f1_Whh, Wf1hh, WN);
    cvt(f2_Wih, Wf2ih, WN); cvt(f2_Whh, Wf2hh, WN);
    cvt(b1_Wih, Wb1ih, WN); cvt(b1_Whh, Wb1hh, WN);
    cvt(b2_Wih, Wb2ih, WN); cvt(b2_Whh, Wb2hh, WN);
    cvt(d1_Wih, Wd1ih, (size_t)G4H * 2 * Hh); cvt(d1_Whh, Wd1hh, WN);
    cvt(d2_Wih, Wd2ih, WN); cvt(d2_Whh, Wd2hh, WN);
    cvt(out_W, WoutT, (size_t)VTt * 2 * Hh);

    // ---- encoder embeddings (S*B rows) ----
    embed_kernel<<<dim3((Ss * Bb * Hh) / 256), dim3(256), 0, stream>>>(src, enc_emb, X16, Ss * Bb);

    // All M dims here are multiples of 32 -> always use the 32x64-per-wave tile.
    auto gemm = [&](const _Float16* A1, const _Float16* W1, int K1,
                    const _Float16* A2, const _Float16* W2, int K2,
                    const float* bias, float* C, int M, int N) {
        int mGroups = M / 32, tN4 = N / 64;
        int waves = mGroups * tN4;
        int blocks = (waves + 7) / 8;
        wmma_gemm_kernel<2><<<dim3(blocks), dim3(256), 0, stream>>>(
            A1, W1, K1, A2, W2, K2, bias, C, mGroups, tN4, N);
    };
    auto gates = [&](const float* g1, const float* g2, const float* bih, const float* bhh,
                     const int* msrc, float* h, float* c, _Float16* h16, float* hstore) {
        lstm_gates_kernel<<<dim3((Bb * Hh) / 256), dim3(256), 0, stream>>>(
            g1, g2, bih, bhh, msrc, h, c, h16, hstore);
    };

    // ---- precompute x @ Wih.T for both layer-1 directions (big WMMA GEMMs) ----
    gemm(X16, Wf1ih, Hh, nullptr, nullptr, 0, nullptr, GpreF, Ss * Bb, G4H);
    gemm(X16, Wb1ih, Hh, nullptr, nullptr, 0, nullptr, GpreB, Ss * Bb, G4H);

    // ---- forward encoder ----
    for (int s = 0; s < Ss; ++s) {
        gemm(h1f16, Wf1hh, Hh, nullptr, nullptr, 0, nullptr, gh1, Bb, G4H);
        gates(GpreF + (size_t)s * Bb * G4H, gh1, f1_bih, f1_bhh, src + s * Bb,
              h1f, c1f, h1f16, nullptr);
        gemm(h1f16, Wf2ih, Hh, h2f16, Wf2hh, Hh, nullptr, gbuf, Bb, G4H);
        gates(gbuf, nullptr, f2_bih, f2_bhh, src + s * Bb,
              h2f, c2f, h2f16, fh + (size_t)s * BH);
    }

    // ---- backward encoder ----
    for (int st = 0; st < Ss; ++st) {
        int s = Ss - 1 - st;
        gemm(h1b16, Wb1hh, Hh, nullptr, nullptr, 0, nullptr, gh1, Bb, G4H);
        gates(GpreB + (size_t)s * Bb * G4H, gh1, b1_bih, b1_bhh, src + s * Bb,
              h1b, c1b, h1b16, nullptr);
        gemm(h1b16, Wb2ih, Hh, h2b16, Wb2hh, Hh, nullptr, gbuf, Bb, G4H);
        gates(gbuf, nullptr, b2_bih, b2_bhh, src + s * Bb,
              h2b, c2b, h2b16, bh + (size_t)s * BH);
    }

    // ---- enc_out = fh + bh/2 ----
    combine_kernel<<<dim3((Ss * Bb * Hh) / 256), dim3(256), 0, stream>>>(
        fh, bh, enc_out, Ss * Bb * Hh);

    // ---- decoder: initial state = final backward states (reuse buffers), ctx = 0 ----
    for (int t = 0; t < Tt; ++t) {
        dec_input_kernel<<<dim3((Bb * 2 * Hh) / 256), dim3(256), 0, stream>>>(
            trg + t * Bb, dec_emb, ctx, x16);
        gemm(x16, Wd1ih, 2 * Hh, h1b16, Wd1hh, Hh, nullptr, gbuf, Bb, G4H);
        gates(gbuf, nullptr, d1_bih, d1_bhh, nullptr, h1b, c1b, h1b16, nullptr);
        gemm(h1b16, Wd2ih, Hh, h2b16, Wd2hh, Hh, nullptr, gbuf, Bb, G4H);
        gates(gbuf, nullptr, d2_bih, d2_bhh, nullptr, h2b, c2b, h2b16, nullptr);
        attn_kernel<<<dim3(Bb), dim3(256), 0, stream>>>(
            enc_out, h2b, src, attn_out + (size_t)t * Bb * Ss, ctx, cat16);
        float* logits_t = logits_out + (size_t)t * Bb * VTt;
        gemm(cat16, WoutT, 2 * Hh, nullptr, nullptr, 0, out_b, logits_t, Bb, VTt);
        argmax_kernel<<<dim3(Bb), dim3(256), 0, stream>>>(logits_t, labels_out + (size_t)t * Bb, VTt);
    }
}
